// IsoGMMLoss_1236950581441
// MI455X (gfx1250) — matrix operand order repaired and checked by reading, not compiled, not run on hardware
//
#include <hip/hip_runtime.h>

typedef __attribute__((ext_vector_type(16))) _Float16 v16h;
typedef __attribute__((ext_vector_type(8)))  float    v8f;

#define NROWS 262144
#define DIM   128
#define KCL   64
#define GRID  2048            // 8 waves/block * 2048 = 16384 waves = nTiles
#define BLOCK 256
#define WAVES_PER_BLOCK (BLOCK / 32)

// workspace layout (bytes):
//   [0,     16384) : mus as f16, row-major [k][d], packed as 4096 uint32 (pairs)
//   [16384, 16640) : m2[64] = ||mu_k||^2 (float)
//   [16640, +8KiB) : per-block partials (GRID floats)
#define WS_MUS_U32   0
#define WS_M2_OFF    16384
#define WS_PART_OFF  16640

static __device__ __forceinline__ float sq4(float4 q) {
  return q.x * q.x + q.y * q.y + q.z * q.z + q.w * q.w;
}

static __device__ __forceinline__ v16h pack16(float4 a, float4 b, float4 c, float4 d) {
  v16h f;
  f[0]  = (_Float16)a.x; f[1]  = (_Float16)a.y; f[2]  = (_Float16)a.z; f[3]  = (_Float16)a.w;
  f[4]  = (_Float16)b.x; f[5]  = (_Float16)b.y; f[6]  = (_Float16)b.z; f[7]  = (_Float16)b.w;
  f[8]  = (_Float16)c.x; f[9]  = (_Float16)c.y; f[10] = (_Float16)c.z; f[11] = (_Float16)c.w;
  f[12] = (_Float16)d.x; f[13] = (_Float16)d.y; f[14] = (_Float16)d.z; f[15] = (_Float16)d.w;
  return f;
}

static __device__ __forceinline__ float wave_sum(float v) {
  v += __shfl_xor(v, 16, 32);
  v += __shfl_xor(v, 8, 32);
  v += __shfl_xor(v, 4, 32);
  v += __shfl_xor(v, 2, 32);
  v += __shfl_xor(v, 1, 32);
  return v;
}

// One-time prep: mus fp32 -> f16 (packed pairs) into ws, plus m2[k] = ||mu_k||^2.
__global__ void __launch_bounds__(256)
iso_gmm_prep(const float* __restrict__ MU, unsigned int* __restrict__ musU,
             float* __restrict__ m2) {
  const int idx = blockIdx.x * 256 + threadIdx.x;  // one uint (2 halfs) each
  if (idx < (KCL * DIM) / 2) {
    _Float16 h0 = (_Float16)MU[2 * idx];
    _Float16 h1 = (_Float16)MU[2 * idx + 1];
    unsigned int u = (unsigned int)__builtin_bit_cast(unsigned short, h0) |
                     ((unsigned int)__builtin_bit_cast(unsigned short, h1) << 16);
    musU[idx] = u;
  }
  if (blockIdx.x == 0 && threadIdx.x < KCL) {
    const float* p = MU + (size_t)threadIdx.x * DIM;
    float s = 0.f;
#pragma unroll
    for (int d = 0; d < DIM; d += 4) s += sq4(*(const float4*)(p + d));
    m2[threadIdx.x] = s;
  }
}

// loss*N = sum_{n,k} r[n,k] * (||x_n||^2 + ||mu_k||^2 - 2 * x_n . mu_k)
// cross = X @ mus^T via v_wmma_f32_16x16x32_f16; ||x||^2 via a 5th WMMA chain:
// (A .* A) @ Ones lands x2 directly in the C/D layout (no cross-lane shuffles).
__global__ void __launch_bounds__(BLOCK)
iso_gmm_main(const float* __restrict__ X, const float* __restrict__ R,
             const unsigned int* __restrict__ musU, const float* __restrict__ m2g,
             float* __restrict__ partial) {
  const int lane = threadIdx.x & 31;
  const int wid  = threadIdx.x >> 5;
  const int half = lane >> 4;   // 0: lanes 0-15, 1: lanes 16-31
  const int mrow = lane & 15;
  const int b8   = half * 16;

  // Stage mus (f16, [k][d] row-major, 16 KB) into LDS with b128 copies.
  __shared__ __align__(16) unsigned int sMus[(KCL * DIM) / 2];
  {
    uint4* s4 = (uint4*)sMus;
    const uint4* g4 = (const uint4*)musU;
    for (int i = threadIdx.x; i < (KCL * DIM) / 8; i += BLOCK) s4[i] = g4[i];
  }
  __syncthreads();

  // ||mu_k||^2 for k = 16t + mrow (C/D layout column), loop-invariant.
  float m2t[4];
#pragma unroll
  for (int t = 0; t < 4; ++t) m2t[t] = m2g[16 * t + mrow];

  // All-ones B fragment for the row-sum (x2) WMMA.
  v16h ones;
#pragma unroll
  for (int i = 0; i < 16; ++i) ones[i] = (_Float16)1.0f;

  // LDS index (uints) of this lane's B-fragment base:
  // B (32x16 f16): lanes 0-15 hold K(d)=d0+0..15, lanes 16-31 hold d0+16..31,
  // column N = k0 + mrow.
  const int bBase = mrow * (DIM / 2) + 8 * half;  // + 16*c + t*16*(DIM/2)

  const int nTiles = NROWS / 16;
  const int gwave  = blockIdx.x * WAVES_PER_BLOCK + wid;

  float acc = 0.f;

  if (gwave < nTiles) {  // exactly one 16-row tile per wave (uniform branch)
    const int n0 = gwave * 16;
    const float* xbase = X + (size_t)(n0 + mrow) * DIM + b8;
    const float* rbase = R + (size_t)(n0 + 8 * half) * KCL + mrow;

    v8f C0 = {}, C1 = {}, C2 = {}, C3 = {}, C4 = {};
#pragma unroll
    for (int c = 0; c < 4; ++c) {
      // A 16x32 f16 as loaded here (same k order as B): half 0 -> K = 0..15 ; half 1 -> K = 16..31
      float4 q0 = *(const float4*)(xbase + 32 * c);
      float4 q1 = *(const float4*)(xbase + 32 * c + 4);
      float4 q2 = *(const float4*)(xbase + 32 * c + 8);
      float4 q3 = *(const float4*)(xbase + 32 * c + 12);
      v16h A  = pack16(q0, q1, q2, q3);
      v16h A2 = A * A;  // v_pk_mul_f16: squared elements for the x2 row-sum
      v16h B0 = *(const v16h*)(sMus + bBase + 16 * c);
      v16h B1 = *(const v16h*)(sMus + bBase + 16 * c + 1 * 16 * (DIM / 2));
      v16h B2 = *(const v16h*)(sMus + bBase + 16 * c + 2 * 16 * (DIM / 2));
      v16h B3 = *(const v16h*)(sMus + bBase + 16 * c + 3 * 16 * (DIM / 2));
      C0 = __builtin_amdgcn_wmma_f32_16x16x32_f16(false, A,  false, B0,   (short)0, C0, false, false);
      C1 = __builtin_amdgcn_wmma_f32_16x16x32_f16(false, A,  false, B1,   (short)0, C1, false, false);
      C2 = __builtin_amdgcn_wmma_f32_16x16x32_f16(false, A,  false, B2,   (short)0, C2, false, false);
      C3 = __builtin_amdgcn_wmma_f32_16x16x32_f16(false, A,  false, B3,   (short)0, C3, false, false);
      C4 = __builtin_amdgcn_wmma_f32_16x16x32_f16(false, A2, false, ones, (short)0, C4, false, false);
    }

    // Fuse with r in the C/D layout: VGPR v -> M = v + 8*half, N = mrow.
    // C4[v] = ||x_{n0+v+8*half}||^2 (replicated across columns).
#pragma unroll
    for (int t = 0; t < 4; ++t) {
      v8f Ct = (t == 0) ? C0 : (t == 1) ? C1 : (t == 2) ? C2 : C3;
#pragma unroll
      for (int v = 0; v < 8; ++v) {
        float rv = rbase[v * KCL + 16 * t];
        acc += rv * (C4[v] + m2t[t] - 2.0f * Ct[v]);
      }
    }
  }

  acc = wave_sum(acc);
  __shared__ float sdata[WAVES_PER_BLOCK];
  if (lane == 0) sdata[wid] = acc;
  __syncthreads();
  if (threadIdx.x == 0) {
    float s = 0.f;
#pragma unroll
    for (int i = 0; i < WAVES_PER_BLOCK; ++i) s += sdata[i];
    partial[blockIdx.x] = s;
  }
}

__global__ void __launch_bounds__(256)
iso_gmm_reduce(const float* __restrict__ partial, float* __restrict__ out) {
  float v = 0.f;
  for (int i = threadIdx.x; i < GRID; i += 256) v += partial[i];
  v = wave_sum(v);
  __shared__ float sdata[256 / 32];
  const int lane = threadIdx.x & 31, wid = threadIdx.x >> 5;
  if (lane == 0) sdata[wid] = v;
  __syncthreads();
  if (threadIdx.x == 0) {
    float s = 0.f;
#pragma unroll
    for (int i = 0; i < 256 / 32; ++i) s += sdata[i];
    out[0] = s * (1.0f / (float)NROWS);
  }
}

extern "C" void kernel_launch(void* const* d_in, const int* in_sizes, int n_in,
                              void* d_out, int out_size, void* d_ws, size_t ws_size,
                              hipStream_t stream) {
  (void)in_sizes; (void)n_in; (void)out_size; (void)ws_size;
  const float* X  = (const float*)d_in[0];
  const float* R  = (const float*)d_in[1];
  const float* MU = (const float*)d_in[2];

  unsigned int* musU = (unsigned int*)((char*)d_ws + WS_MUS_U32);
  float* m2          = (float*)((char*)d_ws + WS_M2_OFF);
  float* partial     = (float*)((char*)d_ws + WS_PART_OFF);
  float* out         = (float*)d_out;

  iso_gmm_prep<<<(KCL * DIM / 2 + 255) / 256, 256, 0, stream>>>(MU, musU, m2);
  iso_gmm_main<<<GRID, BLOCK, 0, stream>>>(X, R, musU, m2, partial);
  iso_gmm_reduce<<<1, 256, 0, stream>>>(partial, out);
}
